// SparseTransformerBlock_26371099197575
// MI455X (gfx1250) — matrix operand order repaired and checked by reading, not compiled, not run on hardware
//
#include <hip/hip_runtime.h>
#include <cstdint>

typedef __bf16 bf16_t;
typedef float  v8f   __attribute__((ext_vector_type(8)));
typedef bf16_t v16bf __attribute__((ext_vector_type(16)));
typedef bf16_t v8bf  __attribute__((ext_vector_type(8)));

#define EPSC   1.1920929e-07f
#define SCALEC 0.125f   // HD^-0.5, HD=64

union FragU { v16bf v; v8bf h[2]; };

// A-fragment: 16-bit A 16x32 layout (lane<16: K 0..7 & 16..23; lane>=16: K 8..15 & 24..31)
__device__ __forceinline__ v16bf fragA(const bf16_t* row, int kbase, int hi) {
  FragU u;
  u.h[0] = *(const v8bf*)(row + kbase + hi * 8);
  u.h[1] = *(const v8bf*)(row + kbase + 16 + hi * 8);
  return u.v;
}
// B-fragment: lane<16 holds K 0..7 & 16..23, lane>=16 holds K 8..15 & 24..31 (column = lane%16), from B^T rows
__device__ __forceinline__ v16bf fragB(const bf16_t* row, int kbase, int hi) {
  FragU u;
  u.h[0] = *(const v8bf*)(row + kbase + hi * 8);
  u.h[1] = *(const v8bf*)(row + kbase + 16 + hi * 8);
  return u.v;
}
__device__ __forceinline__ v8f wmma_bf16(v16bf a, v16bf b, v8f c) {
  return __builtin_amdgcn_wmma_f32_16x16x32_bf16(false, a, false, b, (short)0, c, false, false);
}
__device__ __forceinline__ v8f v8f_zero() {
  v8f z = {0.f, 0.f, 0.f, 0.f, 0.f, 0.f, 0.f, 0.f};
  return z;
}

// CDNA5 async global->LDS copy (ASYNCcnt); lds_off = wave-relative LDS byte offset
// (generic LDS pointers carry the LDS offset in addr[31:0], ISA 10.2).
__device__ __forceinline__ void async_b128(uint32_t lds_off, const void* gptr) {
  asm volatile("global_load_async_to_lds_b128 %0, %1, off"
               :
               : "v"(lds_off), "v"((uint64_t)(uintptr_t)gptr)
               : "memory");
}
__device__ __forceinline__ void wait_async0() {
  asm volatile("s_wait_asynccnt 0x0" ::: "memory");
}
__device__ __forceinline__ uint32_t lds_off32(const void* p) {
  return (uint32_t)(uintptr_t)p;
}

// ---------------------------------------------------------------------------
// fp32 -> bf16 weight convert + transpose: w (K x Nc) row-major -> wt (Nc x K)
// ---------------------------------------------------------------------------
__global__ void convT_k(const float* __restrict__ w, bf16_t* __restrict__ wt, int K, int Nc) {
  long i = (long)blockIdx.x * 256 + threadIdx.x;
  if (i < (long)K * Nc) {
    long kk = i / Nc, n = i % Nc;
    wt[n * (long)K + kk] = (bf16_t)w[i];
  }
}

// ---------------------------------------------------------------------------
// RMSNorm #1: h = feat * rsqrt(mean(feat^2)+eps) * w ; store f32 + bf16
// ---------------------------------------------------------------------------
__global__ __launch_bounds__(256) void rmsnorm1_k(const float* __restrict__ feat,
                                                  const float* __restrict__ w,
                                                  float* __restrict__ hf,
                                                  bf16_t* __restrict__ hb, int N) {
  int row = blockIdx.x * 8 + (threadIdx.x >> 5);
  int lane = threadIdx.x & 31;
  if (row >= N) return;
  const float* x = feat + (long)row * 512;
  float vals[16];
  float ss = 0.f;
#pragma unroll
  for (int i = 0; i < 16; i++) { float t = x[lane + i * 32]; vals[i] = t; ss += t * t; }
#pragma unroll
  for (int off = 16; off >= 1; off >>= 1) ss += __shfl_xor(ss, off, 32);
  float sc = rsqrtf(ss * (1.f / 512.f) + EPSC);
  float* o = hf + (long)row * 512;
  bf16_t* ob = hb + (long)row * 512;
#pragma unroll
  for (int i = 0; i < 16; i++) {
    int c = lane + i * 32;
    float t = vals[i] * sc * w[c];
    o[c] = t;
    ob[c] = (bf16_t)t;
  }
}

// ---------------------------------------------------------------------------
// residual scatter + RMSNorm #2: h2 = h + proj[unpad[i]]; u = rmsnorm(h2)*w (bf16)
// ---------------------------------------------------------------------------
__global__ __launch_bounds__(256) void resid_norm2_k(const float* __restrict__ hf,
                                                     const float* __restrict__ proj,
                                                     const int* __restrict__ unpad,
                                                     const float* __restrict__ w,
                                                     float* __restrict__ h2f,
                                                     bf16_t* __restrict__ ub, int N) {
  int row = blockIdx.x * 8 + (threadIdx.x >> 5);
  int lane = threadIdx.x & 31;
  if (row >= N) return;
  const float* a = hf + (long)row * 512;
  const float* p = proj + (long)unpad[row] * 512;
  float vals[16];
  float ss = 0.f;
#pragma unroll
  for (int i = 0; i < 16; i++) {
    int c = lane + i * 32;
    float t = a[c] + p[c];
    vals[i] = t;
    ss += t * t;
  }
#pragma unroll
  for (int off = 16; off >= 1; off >>= 1) ss += __shfl_xor(ss, off, 32);
  float sc = rsqrtf(ss * (1.f / 512.f) + EPSC);
  float* o = h2f + (long)row * 512;
  bf16_t* ob = ub + (long)row * 512;
#pragma unroll
  for (int i = 0; i < 16; i++) {
    int c = lane + i * 32;
    o[c] = vals[i];
    ob[c] = (bf16_t)(vals[i] * sc * w[c]);
  }
}

// ---------------------------------------------------------------------------
// Generic bf16 GEMM, fp32 accum, double-buffered async global->LDS pipeline.
// C[M x Nc] = A[M x K] (opt. row-gather) * (B^T[Nc x K])^T
// Block tile 64x128x32, 8 waves (4 M x 2 N), each wave 16x64 (4 wmma / k-step).
// EPI: 0 = qkv de-interleave + RoPE -> q/k/v bf16 [b][h][p][d]
//      1 = plain f32 store
//      2 = exact GELU -> bf16
//      3 = residual add -> f32
// ---------------------------------------------------------------------------
template <int EPI>
__global__ __launch_bounds__(256) void gemm_bf16_k(
    const bf16_t* __restrict__ A, const int* __restrict__ gat, const bf16_t* __restrict__ BT,
    int M, int Nc, int K, float* __restrict__ outf, bf16_t* __restrict__ outb,
    bf16_t* __restrict__ qo, bf16_t* __restrict__ ko, bf16_t* __restrict__ vo,
    const float* __restrict__ fcos, const float* __restrict__ fsin, const int* __restrict__ pad,
    const float* __restrict__ resid) {
  __shared__ __attribute__((aligned(16))) bf16_t As[2][64][40];
  __shared__ __attribute__((aligned(16))) bf16_t Bs[2][128][40];
  const int tid = threadIdx.x;
  const int wave = tid >> 5, lane = tid & 31;
  const int l16 = lane & 15, hi = lane >> 4;
  const int wm = wave & 3, wn = wave >> 2;
  const int mblk = blockIdx.x * 64;
  const int nblk = blockIdx.y * 128;

  v8f acc[4];
#pragma unroll
  for (int j = 0; j < 4; j++) acc[j] = v8f_zero();

  const int arow = tid >> 2, akoff = (tid & 3) * 8;   // A: 64 rows x 32 k, 8 elems/thread
  const int brow = tid >> 1, bkoff = (tid & 1) * 16;  // B: 128 rows x 32 k, 16 elems/thread
  int gm = mblk + arow;
  if (gm >= M) gm = 0;
  const long grow = gat ? (long)gat[gm] : (long)gm;
  const bf16_t* aptr = A + grow * (long)K + akoff;
  const bf16_t* bptr = BT + (long)(nblk + brow) * K + bkoff;

  const uint32_t laA[2] = {lds_off32(&As[0][arow][akoff]), lds_off32(&As[1][arow][akoff])};
  const uint32_t laB[2] = {lds_off32(&Bs[0][brow][bkoff]), lds_off32(&Bs[1][brow][bkoff])};

  // prologue: async-copy tile 0 into buffer 0
  async_b128(laA[0], aptr);
  async_b128(laB[0], bptr);
  async_b128(laB[0] + 16, bptr + 8);

  const int nk = K >> 5;
  for (int i = 0; i < nk; i++) {
    wait_async0();      // this wave's async stores to buf (i&1) have landed in LDS
    __syncthreads();    // publish; also guarantees buf ((i+1)&1) is no longer being read
    if (i + 1 < nk) {   // async-prefetch tile i+1 into the other buffer
      const int nb2 = (i + 1) & 1;
      const int k1 = (i + 1) << 5;
      async_b128(laA[nb2], aptr + k1);
      async_b128(laB[nb2], bptr + k1);
      async_b128(laB[nb2] + 16, bptr + k1 + 8);
    }
    const int buf = i & 1;
    v16bf af = fragA(&As[buf][wm * 16 + l16][0], 0, hi);
#pragma unroll
    for (int j = 0; j < 4; j++) {
      v16bf bfr = fragB(&Bs[buf][wn * 64 + j * 16 + l16][0], 0, hi);
      acc[j] = wmma_bf16(af, bfr, acc[j]);
    }
  }

  // Epilogue. C layout: elem r of v8f -> row (r + 8*hi), col (lane%16) within 16x16 tile.
#pragma unroll
  for (int j = 0; j < 4; j++) {
#pragma unroll
    for (int r = 0; r < 8; r++) {
      const float val = acc[j][r];
      const int row = mblk + wm * 16 + r + 8 * hi;
      const int col = nblk + wn * 64 + j * 16 + l16;
      if constexpr (EPI == 0) {
        // adjacent lanes hold adjacent d -> exchange RoPE partner
        const float partner = __shfl_xor(val, 1, 32);
        const int bb = row >> 7, rl = row & 127;
        const int c1 = col >> 9, c0 = col & 511;
        const int jf = 3 * rl + c1;          // qkv.reshape(-1,3,128,8,64) de-interleave
        const int s = jf >> 7, pp = jf & 127;
        const int hh = c0 >> 6, dd = c0 & 63;
        const long didx = (((long)(bb * 8 + hh)) * 128 + pp) * 64 + dd;
        if (s == 2) {
          vo[didx] = (bf16_t)val;
        } else {
          const int pos = pad[bb * 128 + pp];
          const float cs = fcos[(long)pos * 32 + (dd >> 1)];
          const float sn = fsin[(long)pos * 32 + (dd >> 1)];
          float ov = ((dd & 1) == 0) ? (val * cs - partner * sn) : (partner * sn + val * cs);
          if (s == 0) {
            qo[didx] = (bf16_t)(ov * SCALEC);
          } else {
            ko[didx] = (bf16_t)ov;
          }
        }
      } else if constexpr (EPI == 1) {
        if (row < M) outf[(long)row * Nc + col] = val;
      } else if constexpr (EPI == 2) {
        if (row < M) {
          const float g = 0.5f * val * (1.f + erff(val * 0.70710678118654752f));
          outb[(long)row * Nc + col] = (bf16_t)g;
        }
      } else {
        if (row < M) outf[(long)row * Nc + col] = resid[(long)row * Nc + col] + val;
      }
    }
  }
}

// ---------------------------------------------------------------------------
// Block-diagonal attention: one workgroup per (block b, head h).
// S = softmax(Qs . K^T) (Q pre-scaled), O = S . V.  All matmuls via bf16 WMMA.
// Q/K staged via async global->LDS; V vector-loaded and transposed into LDS.
// ---------------------------------------------------------------------------
__global__ __launch_bounds__(256) void attn_k(const bf16_t* __restrict__ q,
                                              const bf16_t* __restrict__ k,
                                              const bf16_t* __restrict__ v,
                                              bf16_t* __restrict__ o, int nb) {
  __shared__ __attribute__((aligned(16))) bf16_t Qs[128][72];
  __shared__ __attribute__((aligned(16))) bf16_t Ks[128][72];
  __shared__ __attribute__((aligned(16))) bf16_t Vt[64][136];   // V transposed (d x p)
  __shared__ __attribute__((aligned(16))) bf16_t Ss[128][136];  // softmax probs, bf16
  const int b = blockIdx.x, h = blockIdx.y;
  const long base = (((long)b * 8 + h) * 128) * 64;
  const int tid = threadIdx.x, wave = tid >> 5, lane = tid & 31;
  const int l16 = lane & 15, hi = lane >> 4;

  {
    const int row = tid >> 1, half = (tid & 1) * 32;
    const bf16_t* qp = q + base + row * 64 + half;
    const bf16_t* kp = k + base + row * 64 + half;
    const bf16_t* vp = v + base + row * 64 + half;
    const uint32_t lq = lds_off32(&Qs[row][half]);
    const uint32_t lk = lds_off32(&Ks[row][half]);
#pragma unroll
    for (int i = 0; i < 4; i++) {
      async_b128(lq + i * 16, qp + i * 8);
      async_b128(lk + i * 16, kp + i * 8);
    }
    v8bf vv[4];
#pragma unroll
    for (int i = 0; i < 4; i++) vv[i] = *(const v8bf*)(vp + i * 8);
#pragma unroll
    for (int i = 0; i < 4; i++)
#pragma unroll
      for (int e = 0; e < 8; e++) Vt[half + i * 8 + e][row] = vv[i][e];
  }
  wait_async0();
  __syncthreads();

  // Phase 1: S(16x128 per wave) = Q . K^T   (B = K^T taken from row-major Ks rows)
  v8f sacc[8];
#pragma unroll
  for (int j = 0; j < 8; j++) sacc[j] = v8f_zero();
#pragma unroll
  for (int kk = 0; kk < 64; kk += 32) {
    v16bf af = fragA(&Qs[wave * 16 + l16][0], kk, hi);
#pragma unroll
    for (int j = 0; j < 8; j++) {
      v16bf bfr = fragB(&Ks[j * 16 + l16][0], kk, hi);
      sacc[j] = wmma_bf16(af, bfr, sacc[j]);
    }
  }

  // softmax per row; lanes 0..15 and 16..31 hold independent rows -> half-wave reductions
#pragma unroll
  for (int r = 0; r < 8; r++) {
    float mx = -3.4e38f;
#pragma unroll
    for (int j = 0; j < 8; j++) mx = fmaxf(mx, sacc[j][r]);
#pragma unroll
    for (int off = 1; off < 16; off <<= 1) mx = fmaxf(mx, __shfl_xor(mx, off, 32));
    float e[8];
    float sum = 0.f;
#pragma unroll
    for (int j = 0; j < 8; j++) { e[j] = expf(sacc[j][r] - mx); sum += e[j]; }
#pragma unroll
    for (int off = 1; off < 16; off <<= 1) sum += __shfl_xor(sum, off, 32);
    const float inv = 1.f / sum;
    const int row = wave * 16 + r + 8 * hi;
#pragma unroll
    for (int j = 0; j < 8; j++) Ss[row][j * 16 + l16] = (bf16_t)(e[j] * inv);
  }
  __syncthreads();

  // Phase 2: O(16x64 per wave) = S . V   (B from Vt rows)
  v8f oacc[4];
#pragma unroll
  for (int j = 0; j < 4; j++) oacc[j] = v8f_zero();
#pragma unroll
  for (int kk = 0; kk < 128; kk += 32) {
    v16bf af = fragA(&Ss[wave * 16 + l16][0], kk, hi);
#pragma unroll
    for (int j = 0; j < 4; j++) {
      v16bf bfr = fragB(&Vt[j * 16 + l16][0], kk, hi);
      oacc[j] = wmma_bf16(af, bfr, oacc[j]);
    }
  }
  // store (b,p,h,d) -> row b*128+p, col h*64+d of (Np x 512)
#pragma unroll
  for (int j = 0; j < 4; j++) {
#pragma unroll
    for (int r = 0; r < 8; r++) {
      const int p = wave * 16 + r + 8 * hi;
      o[((long)(b * 128 + p)) * 512 + h * 64 + j * 16 + l16] = (bf16_t)oacc[j][r];
    }
  }
}

// ---------------------------------------------------------------------------
extern "C" void kernel_launch(void* const* d_in, const int* in_sizes, int n_in, void* d_out,
                              int out_size, void* d_ws, size_t ws_size, hipStream_t stream) {
  (void)n_in; (void)out_size; (void)ws_size;
  const float* feat  = (const float*)d_in[0];
  const float* fcos  = (const float*)d_in[1];
  const float* fsin  = (const float*)d_in[2];
  const int*   pad   = (const int*)d_in[3];
  const int*   unpad = (const int*)d_in[4];
  const float* n1w   = (const float*)d_in[5];
  const float* n2w   = (const float*)d_in[6];
  const float* wqkv  = (const float*)d_in[7];
  const float* wproj = (const float*)d_in[8];
  const float* wmlp1 = (const float*)d_in[9];
  const float* wmlp2 = (const float*)d_in[10];

  const int N  = in_sizes[0] / 512;  // 65471
  const int Np = in_sizes[3];        // 65664
  const int nb = Np / 128;           // 513

  char* p = (char*)d_ws;
  auto alloc = [&](size_t bytes) -> char* {
    char* r = p;
    p += (bytes + 255) & ~(size_t)255;
    return r;
  };
  float*  h_f32   = (float*)alloc((size_t)N * 512 * 4);
  float*  h2_f32  = (float*)alloc((size_t)N * 512 * 4);
  bf16_t* h_bf    = (bf16_t*)alloc((size_t)N * 512 * 2);
  bf16_t* u_bf    = (bf16_t*)alloc((size_t)N * 512 * 2);
  bf16_t* wqkvT   = (bf16_t*)alloc((size_t)1536 * 512 * 2);
  bf16_t* wprojT  = (bf16_t*)alloc((size_t)512 * 512 * 2);
  bf16_t* wmlp1T  = (bf16_t*)alloc((size_t)2048 * 512 * 2);
  bf16_t* wmlp2T  = (bf16_t*)alloc((size_t)512 * 2048 * 2);
  float*  proj32  = (float*)alloc((size_t)Np * 512 * 4);
  bf16_t* qb      = (bf16_t*)alloc((size_t)Np * 512 * 2);
  bf16_t* kb      = (bf16_t*)alloc((size_t)Np * 512 * 2);
  bf16_t* vb      = (bf16_t*)alloc((size_t)Np * 512 * 2);
  bf16_t* ab      = (bf16_t*)alloc((size_t)Np * 512 * 2);
  // GELU activation (N x 2048 bf16) aliases q/k/v/attn region, which is dead by then.
  bf16_t* act = qb;

  // weight convert+transpose
  convT_k<<<(512 * 1536 + 255) / 256, 256, 0, stream>>>(wqkv, wqkvT, 512, 1536);
  convT_k<<<(512 * 512 + 255) / 256, 256, 0, stream>>>(wproj, wprojT, 512, 512);
  convT_k<<<(512 * 2048 + 255) / 256, 256, 0, stream>>>(wmlp1, wmlp1T, 512, 2048);
  convT_k<<<(2048 * 512 + 255) / 256, 256, 0, stream>>>(wmlp2, wmlp2T, 2048, 512);

  // rmsnorm1
  rmsnorm1_k<<<(N + 7) / 8, 256, 0, stream>>>(feat, n1w, h_f32, h_bf, N);

  // qkv GEMM (gathered A via pad) + de-interleave + RoPE -> q/k/v
  gemm_bf16_k<0><<<dim3((Np + 63) / 64, 1536 / 128), 256, 0, stream>>>(
      h_bf, pad, wqkvT, Np, 1536, 512, nullptr, nullptr, qb, kb, vb, fcos, fsin, pad, nullptr);

  // attention per (block, head)
  attn_k<<<dim3(nb, 8), 256, 0, stream>>>(qb, kb, vb, ab, nb);

  // proj GEMM -> f32
  gemm_bf16_k<1><<<dim3((Np + 63) / 64, 512 / 128), 256, 0, stream>>>(
      ab, nullptr, wprojT, Np, 512, 512, proj32, nullptr, nullptr, nullptr, nullptr, nullptr,
      nullptr, nullptr, nullptr);

  // residual (scatter via unpad) + rmsnorm2
  resid_norm2_k<<<(N + 7) / 8, 256, 0, stream>>>(h_f32, proj32, unpad, n2w, h2_f32, u_bf, N);

  // mlp1 GEMM + GELU -> bf16
  gemm_bf16_k<2><<<dim3((N + 63) / 64, 2048 / 128), 256, 0, stream>>>(
      u_bf, nullptr, wmlp1T, N, 2048, 512, nullptr, act, nullptr, nullptr, nullptr, nullptr,
      nullptr, nullptr, nullptr);

  // mlp2 GEMM + residual -> d_out (f32)
  gemm_bf16_k<3><<<dim3((N + 63) / 64, 512 / 128), 256, 0, stream>>>(
      act, nullptr, wmlp2T, N, 512, 2048, (float*)d_out, nullptr, nullptr, nullptr, nullptr,
      nullptr, nullptr, nullptr, h2_f32);
}
